// NTXentLoss_2508260901698
// MI455X (gfx1250) — matrix operand order repaired and checked by reading, not compiled, not run on hardware
//
#include <hip/hip_runtime.h>
#include <hip/hip_bf16.h>
#include <math.h>

typedef __attribute__((ext_vector_type(16))) _Float16 v16h;
typedef __attribute__((ext_vector_type(8)))  float    v8f;

#define D_DIM 512
#define DW_PER_ROW (D_DIM / 2)   // 256 dwords (packed f16 pairs) per row
#define TEMP_INV 2.0f            // 1 / temperature (0.5)
#define LSE_SHIFT 2.0f           // logits bounded: x = 2*cos <= 2 -> fixed-shift LSE

union FragU { v16h h; unsigned int u[8]; };

// ---------------------------------------------------------------------------
// Kernel 1: row-wise L2 normalize z = [z_i; z_j] into an f16 workspace matrix,
// and zero the scalar output accumulator (d_out is poisoned by the harness).
// One 256-thread block per row; each thread owns 2 of the 512 columns.
// ---------------------------------------------------------------------------
__global__ __launch_bounds__(256)
void ntxent_normalize(const float* __restrict__ zi, const float* __restrict__ zj,
                      _Float16* __restrict__ zn, float* __restrict__ out, int B) {
    const int row = blockIdx.x;
    const int tid = threadIdx.x;
    const float* src = (row < B) ? (zi + (size_t)row * D_DIM)
                                 : (zj + (size_t)(row - B) * D_DIM);
    const float x0 = src[tid];
    const float x1 = src[tid + 256];
    float ss = x0 * x0 + x1 * x1;
    #pragma unroll
    for (int d = 16; d > 0; d >>= 1) ss += __shfl_xor(ss, d, 32);

    __shared__ float wsum[8];
    const int lane = tid & 31, wid = tid >> 5;
    if (lane == 0) wsum[wid] = ss;
    __syncthreads();
    float tot = 0.f;
    #pragma unroll
    for (int w = 0; w < 8; ++w) tot += wsum[w];

    const float scale = 1.0f / fmaxf(sqrtf(tot), 1e-8f);
    zn[(size_t)row * D_DIM + tid]       = (_Float16)(x0 * scale);
    zn[(size_t)row * D_DIM + tid + 256] = (_Float16)(x1 * scale);
    if (row == 0 && tid == 0) out[0] = 0.0f;
}

// ---------------------------------------------------------------------------
// Kernel 2: streaming WMMA GEMM (zn @ zn^T) with fixed-shift logsumexp.
// Block = 256 threads = 8 waves arranged 2(M) x 4(N). Each wave computes a
// 16x32 C strip (two 16x16 tiles sharing the A fragment), so the block covers
// 32 rows x 128 cols per j-step, streaming j over N. Since all logits are
// bounded by 2 (unit rows, T=0.5), logsumexp uses the fixed shift M=2:
//   lse_i = log( sum_{j != i} exp(sim_ij - 2) ) + 2
// which makes the hot loop fully branchless (v_cndmask masks, pure sums).
// ---------------------------------------------------------------------------
__global__ __launch_bounds__(256)
void ntxent_loss(const _Float16* __restrict__ zn, float* __restrict__ out, int B) {
    const int N     = 2 * B;
    const int tid   = threadIdx.x;
    const int lane  = tid & 31;
    const int wid   = tid >> 5;
    const int waveM = wid & 1;   // 0..1 -> which 16-row slab
    const int waveN = wid >> 1;  // 0..3 -> which 32-col slab within a j-step
    const int h     = lane >> 4; // half-wave id
    const int nl    = lane & 15;
    const int mBase = blockIdx.x * 32 + waveM * 16;

    const unsigned int* zn32 = (const unsigned int*)zn;
    // A-matrix (16x32 f16): lane holds row M = lane%16; packed K pairs.
    const unsigned int* aPtr = zn32 + (size_t)(mBase + nl) * DW_PER_ROW;

    // Per-lane dword offsets within one 32-deep K step (contiguous K order, same for A and B):
    //  A: K = 16*h + 2*v + e                 ->  dword 8*h + v
    //  B: K = 16*h + 2*v + e                 ->  dword 8*h + v
    int aOff[8], bOff[8];
    #pragma unroll
    for (int v = 0; v < 8; ++v) {
        aOff[v] = 8 * h + v;
        bOff[v] = 8 * h + v;
    }

    // Per-lane row ids and positive-column ids for the 8 C rows.
    int mGlob[8], posCol[8];
    #pragma unroll
    for (int r = 0; r < 8; ++r) {
        mGlob[r]  = mBase + r + 8 * h;              // C layout: M = r + 8*half
        posCol[r] = (mGlob[r] < B) ? (mGlob[r] + B) : (mGlob[r] - B);
    }

    float sum[8], pos[8];
    #pragma unroll
    for (int r = 0; r < 8; ++r) { sum[r] = 0.f; pos[r] = 0.f; }

    for (int jb = waveN * 32; jb < N; jb += 128) {
        // B-matrix (32x16 f16): lane holds column N = lane%16 == zn row jb+nl.
        const unsigned int* bPtr0 = zn32 + (size_t)(jb + nl) * DW_PER_ROW;
        const unsigned int* bPtr1 = bPtr0 + (size_t)16 * DW_PER_ROW;
        v8f acc0 = {}, acc1 = {};
        #pragma unroll 4
        for (int kd = 0; kd < DW_PER_ROW; kd += 16) {
            FragU a, b0, b1;
            #pragma unroll
            for (int v = 0; v < 8; ++v) {
                a.u[v]  = aPtr[kd + aOff[v]];
                b0.u[v] = bPtr0[kd + bOff[v]];
                b1.u[v] = bPtr1[kd + bOff[v]];
            }
            acc0 = __builtin_amdgcn_wmma_f32_16x16x32_f16(
                       false, a.h, false, b0.h, (short)0, acc0, false, false);
            acc1 = __builtin_amdgcn_wmma_f32_16x16x32_f16(
                       false, a.h, false, b1.h, (short)0, acc1, false, false);
        }

        const int n0 = jb + nl;        // global column of tile 0 for this lane
        const int n1 = n0 + 16;        // global column of tile 1
        #pragma unroll
        for (int r = 0; r < 8; ++r) {
            const float x0 = TEMP_INV * acc0[r];
            const float x1 = TEMP_INV * acc1[r];
            const float e0 = __expf(x0 - LSE_SHIFT);
            const float e1 = __expf(x1 - LSE_SHIFT);
            sum[r] += (n0 != mGlob[r]) ? e0 : 0.f;   // mask self-similarity
            sum[r] += (n1 != mGlob[r]) ? e1 : 0.f;
            pos[r] += (n0 == posCol[r]) ? x0 : 0.f;  // capture positive logit
            pos[r] += (n1 == posCol[r]) ? x1 : 0.f;
        }
    }

    __shared__ float lds_s[32][4];
    __shared__ float lds_p[32][4];

    // Reduce the 16 lanes of each half (they share M per r, differ in N).
    #pragma unroll
    for (int r = 0; r < 8; ++r) {
        float ssum = sum[r], pp = pos[r];
        #pragma unroll
        for (int d = 1; d < 16; d <<= 1) {
            ssum += __shfl_xor(ssum, d, 32);
            pp   += __shfl_xor(pp, d, 32);
        }
        if (nl == 0) {
            const int rowLocal = waveM * 16 + r + 8 * h;  // 0..31
            lds_s[rowLocal][waveN] = ssum;
            lds_p[rowLocal][waveN] = pp;
        }
    }
    __syncthreads();

    // Merge the 4 column-wave partials per row; accumulate scalar loss.
    if (tid < 32) {
        float ssum = 0.f, pp = 0.f;
        #pragma unroll
        for (int wv = 0; wv < 4; ++wv) {
            ssum += lds_s[tid][wv];
            pp   += lds_p[tid][wv];
        }
        float lossRow = __logf(ssum) + LSE_SHIFT - pp;   // logsumexp - positive
        #pragma unroll
        for (int d = 16; d > 0; d >>= 1) lossRow += __shfl_xor(lossRow, d, 32);
        if (tid == 0) atomicAdd(out, lossRow / (float)N);
    }
}

// ---------------------------------------------------------------------------
extern "C" void kernel_launch(void* const* d_in, const int* in_sizes, int n_in,
                              void* d_out, int out_size, void* d_ws, size_t ws_size,
                              hipStream_t stream) {
    (void)n_in; (void)out_size; (void)ws_size;
    const float* zi = (const float*)d_in[0];
    const float* zj = (const float*)d_in[1];
    float* out = (float*)d_out;
    const int B = in_sizes[0] / D_DIM;   // 2048
    const int N = 2 * B;                 // 4096
    _Float16* zn = (_Float16*)d_ws;      // N * D f16 = 4 MB scratch

    ntxent_normalize<<<N, 256, 0, stream>>>(zi, zj, zn, out, B);
    ntxent_loss<<<N / 32, 256, 0, stream>>>(zn, out, B);
}
